// CNModel_85856396248063
// MI455X (gfx1250) — matrix operand order repaired and checked by reading, not compiled, not run on hardware
//
#include <hip/hip_runtime.h>
#include <math.h>

// ---------------------------------------------------------------------------
// CNModel: agg = scatter_add(x[src] -> dst); h = relu(agg @ W);
//          out = sigmoid(h^T @ h)
// MI455X / gfx1250 (wave32):
//  - scatter-add in f32 via global_atomic_add_f32 (agg staged in d_out; both
//    x and agg are 64MB -> fully resident in the 192MB L2)
//  - GEMMs via v_wmma_f32_16x16x32_bf16 (bf16 in, f32 accumulate): ~8x the
//    per-instruction math of the f32 WMMA path for ~200 GFLOP of GEMM work
//  - GEMM1 stores relu(h) *transposed* in bf16 (ht): the 16x16 f32 D-fragment
//    (N=lane%16, M=8*(lane/16)+r) makes the transposed store a single
//    contiguous 16B store per lane, and ht row-major is exactly what GEMM2
//    needs for BOTH its A and B fragments.
//  - GEMM2 exploits symmetry of h^T h: triangular 64x64 wave-tiles, mirror
//    writes vectorized (per-lane-contiguous v8f stores).
// ---------------------------------------------------------------------------

typedef __attribute__((ext_vector_type(16))) __bf16          v16bf;
typedef __attribute__((ext_vector_type(8)))  __bf16          v8bf;
typedef __attribute__((ext_vector_type(2)))  __bf16          v2bf;
typedef __attribute__((ext_vector_type(8)))  float           v8f;
typedef __attribute__((ext_vector_type(8)))  unsigned short  v8u;

#define NDIM 4096

__device__ __forceinline__ unsigned short f32_to_bf16_rne(float f) {
    unsigned int u = __builtin_bit_cast(unsigned int, f);
    u += 0x7FFFu + ((u >> 16) & 1u);     // round to nearest even
    return (unsigned short)(u >> 16);
}

// pack two f32 -> two bf16 (one v_cvt_pk_bf16_f32 when available)
__device__ __forceinline__ unsigned int pack_bf16x2(float a, float b) {
#if __has_builtin(__builtin_amdgcn_cvt_pk_bf16_f32)
    v2bf r = __builtin_amdgcn_cvt_pk_bf16_f32(a, b);
    return __builtin_bit_cast(unsigned int, r);
#else
    return (unsigned int)f32_to_bf16_rne(a) |
           ((unsigned int)f32_to_bf16_rne(b) << 16);
#endif
}

// ---------------- zero init (f32, vectorized) ----------------
__global__ void zero_f32_kernel(float* __restrict__ p, long long n4) {
    long long i = (long long)blockIdx.x * blockDim.x + threadIdx.x;
    float4* p4 = (float4*)p;
    long long stride = (long long)gridDim.x * blockDim.x;
    for (long long j = i; j < n4; j += stride)
        p4[j] = make_float4(0.f, 0.f, 0.f, 0.f);
}

// ---------------- scatter add: agg[dst[e]][c] += x[src[e]][c] ----------------
__global__ void scatter_add_kernel(const float* __restrict__ x,
                                   const int*   __restrict__ src,
                                   const int*   __restrict__ dst,
                                   float* __restrict__ agg,
                                   int E) {
    long long gid = (long long)blockIdx.x * blockDim.x + threadIdx.x;
    long long total = (long long)E * NDIM;
    if (gid >= total) return;
    int e = (int)(gid >> 12);          // NDIM = 4096 = 2^12 columns
    int c = (int)(gid & (NDIM - 1));
    int s = src[e];
    int d = dst[e];
    atomicAdd(agg + (size_t)d * NDIM + c, x[(size_t)s * NDIM + c]);
}

// ---------------- f32 -> bf16 elementwise (4 at a time) ----------------
__global__ void cvt_bf16_kernel(const float* __restrict__ in,
                                unsigned short* __restrict__ out,
                                long long n4) {
    long long i = (long long)blockIdx.x * blockDim.x + threadIdx.x;
    if (i >= n4) return;
    float4 v = ((const float4*)in)[i];
    uint2 o;
    o.x = pack_bf16x2(v.x, v.y);
    o.y = pack_bf16x2(v.z, v.w);
    ((uint2*)out)[i] = o;
}

// ---------------- transpose + convert: out[c][r] = bf16(in[r][c]) ----------------
__global__ void transpose_bf16_kernel(const float* __restrict__ in,
                                      unsigned short* __restrict__ out) {
    __shared__ float tile[32][33];
    int x0 = blockIdx.x * 32;
    int y0 = blockIdx.y * 32;
    int tx = threadIdx.x & 31;
    int ty = threadIdx.x >> 5;          // 0..7 (256 threads)
    #pragma unroll
    for (int r = ty; r < 32; r += 8)
        tile[r][tx] = in[(size_t)(y0 + r) * NDIM + x0 + tx];
    __syncthreads();
    #pragma unroll
    for (int r = ty; r < 32; r += 8)
        out[(size_t)(x0 + r) * NDIM + y0 + tx] = f32_to_bf16_rne(tile[tx][r]);
}

// ---------------- fragment loaders (contiguous-k layouts, wave32) ----------------
// A fragment, 16-bit 16x32: lane half h: elems 0..15 -> K = kb+16h+0..15 (contiguous)
__device__ __forceinline__ v16bf load_frag_a(const unsigned short* __restrict__ base,
                                             int row, int kb, int half) {
    return *(const v16bf*)(base + (size_t)row * NDIM + kb + half * 16);
}
// B fragment, 16-bit 32x16 from a transposed (column-major) matrix BT[n][k]:
// lane = column n, elems 0..15 -> K = kb + 16*half + 0..15 (contiguous)
__device__ __forceinline__ v16bf load_frag_b(const unsigned short* __restrict__ baseT,
                                             int col, int kb, int half) {
    return *(const v16bf*)(baseT + (size_t)col * NDIM + kb + half * 16);
}

// ---------------- GEMM1: ht = relu(aggb @ W)^T  (bf16 out, transposed) --------
__global__ __launch_bounds__(256, 1)
void gemm1_wmma_kernel(const unsigned short* __restrict__ A,   // aggb [N][N] row-major
                       const unsigned short* __restrict__ BT,  // W^T  [N][N] (BT[n][k]=W[k][n])
                       unsigned short* __restrict__ HT) {      // ht   [N][N] (HT[n][m]=relu(h[m][n]))
    const int wid  = blockIdx.x * (blockDim.x >> 5) + (threadIdx.x >> 5);
    const int lane = threadIdx.x & 31;
    const int TPD  = NDIM / 64;            // 64 wave-tiles per dim
    const int mBase = (wid / TPD) * 64;
    const int nBase = (wid % TPD) * 64;
    const int ln   = lane & 15;
    const int half = lane >> 4;

    v8f acc[4][4] = {};
    for (int kb = 0; kb < NDIM; kb += 32) {
        v16bf a[4], b[4];
        #pragma unroll
        for (int i = 0; i < 4; ++i) {
            a[i] = load_frag_a(A,  mBase + i * 16 + ln, kb, half);
            b[i] = load_frag_b(BT, nBase + i * 16 + ln, kb, half);
        }
        #pragma unroll
        for (int i = 0; i < 4; ++i)
            #pragma unroll
            for (int j = 0; j < 4; ++j)
                acc[i][j] = __builtin_amdgcn_wmma_f32_16x16x32_bf16(
                    false, a[i], false, b[j], (short)0, acc[i][j], false, false);
    }
    // D layout: N = ln, M = 8*half + r  ->  HT[n][m] store is contiguous over r.
    #pragma unroll
    for (int i = 0; i < 4; ++i)
        #pragma unroll
        for (int j = 0; j < 4; ++j) {
            int n  = nBase + j * 16 + ln;
            int m0 = mBase + i * 16 + half * 8;
            unsigned int pk[4];
            #pragma unroll
            for (int r = 0; r < 4; ++r) {
                float f0 = acc[i][j][2 * r];     f0 = f0 > 0.f ? f0 : 0.f;
                float f1 = acc[i][j][2 * r + 1]; f1 = f1 > 0.f ? f1 : 0.f;
                pk[r] = pack_bf16x2(f0, f1);
            }
            uint4 o = make_uint4(pk[0], pk[1], pk[2], pk[3]);
            *(uint4*)(HT + (size_t)n * NDIM + m0) = o;
        }
}

// ---------------- GEMM2: out = sigmoid(h^T h), symmetric upper-tri tiles -----
__global__ __launch_bounds__(256, 1)
void gemm2_wmma_kernel(const unsigned short* __restrict__ HT,  // ht [N][N]
                       float* __restrict__ out) {
    const int wid  = blockIdx.x * (blockDim.x >> 5) + (threadIdx.x >> 5);
    const int lane = threadIdx.x & 31;
    const int T    = NDIM / 64;            // 64 tiles/dim; pairs = T(T+1)/2 = 2080
    // decode triangular pair (ti <= tj): f(i) = i*T - i*(i-1)/2
    float tf = (2.f * T + 1.f);
    int ti = (int)((tf - sqrtf(tf * tf - 8.f * (float)wid)) * 0.5f);
    if (ti < 0) ti = 0;
    if (ti > T - 1) ti = T - 1;
    while (ti > 0 && wid < ti * T - (ti * (ti - 1)) / 2) --ti;
    while (wid >= (ti + 1) * T - ((ti + 1) * ti) / 2) ++ti;
    const int tj = ti + (wid - (ti * T - (ti * (ti - 1)) / 2));
    const int mBase = ti * 64;
    const int nBase = tj * 64;
    const int ln   = lane & 15;
    const int half = lane >> 4;

    v8f acc[4][4] = {};
    for (int kb = 0; kb < NDIM; kb += 32) {
        v16bf a[4], b[4];
        #pragma unroll
        for (int i = 0; i < 4; ++i) {
            // pred[m][n] = sum_k ht[m][k]*ht[n][k]: A rows & B columns both from HT
            a[i] = load_frag_a(HT, mBase + i * 16 + ln, kb, half);
            b[i] = load_frag_b(HT, nBase + i * 16 + ln, kb, half);
        }
        #pragma unroll
        for (int i = 0; i < 4; ++i)
            #pragma unroll
            for (int j = 0; j < 4; ++j)
                acc[i][j] = __builtin_amdgcn_wmma_f32_16x16x32_bf16(
                    false, a[i], false, b[j], (short)0, acc[i][j], false, false);
    }
    #pragma unroll
    for (int i = 0; i < 4; ++i)
        #pragma unroll
        for (int j = 0; j < 4; ++j) {
            int n  = nBase + j * 16 + ln;
            int m0 = mBase + i * 16 + half * 8;
            v8f s;
            #pragma unroll
            for (int r = 0; r < 8; ++r)
                s[r] = 1.0f / (1.0f + expf(-acc[i][j][r]));
            // direct (m, n): coalesced across lanes (n = ln varies)
            #pragma unroll
            for (int r = 0; r < 8; ++r)
                out[(size_t)(m0 + r) * NDIM + n] = s[r];
            // mirror (n, m): per-lane contiguous 32B run -> one vector store
            *(v8f*)(out + (size_t)n * NDIM + m0) = s;  // diag tiles: benign dup
        }
}

// ---------------------------------------------------------------------------
extern "C" void kernel_launch(void* const* d_in, const int* in_sizes, int n_in,
                              void* d_out, int out_size, void* d_ws, size_t ws_size,
                              hipStream_t stream) {
    const float* x  = (const float*)d_in[0];
    const int*   ei = (const int*)d_in[1];          // [2, E] : src then dst
    const float* W  = (const float*)d_in[2];
    const int E = in_sizes[1] / 2;
    const int* src = ei;
    const int* dst = ei + E;

    const long long NN = (long long)NDIM * NDIM;

    // d_out doubles as f32 agg staging buffer (overwritten by GEMM2 at the end)
    float* agg = (float*)d_out;

    // workspace: aggb (bf16, 32MB) | Wt (bf16, 32MB) | ht (bf16, 32MB)
    unsigned short* aggb = (unsigned short*)d_ws;
    unsigned short* wtb  = aggb + NN;
    unsigned short* ht   = wtb + NN;

    // 1) agg = 0
    {
        long long n4 = NN / 4;
        int blocks = (int)((n4 + 255) / 256);
        if (blocks > 16384) blocks = 16384;
        zero_f32_kernel<<<blocks, 256, 0, stream>>>(agg, n4);
    }
    // 2) agg[dst] += x[src]   (f32 atomics, L2-resident target)
    {
        long long total = (long long)E * NDIM;
        int blocks = (int)((total + 255) / 256);
        scatter_add_kernel<<<blocks, 256, 0, stream>>>(x, src, dst, agg, E);
    }
    // 3) aggb = bf16(agg)
    {
        long long n4 = NN / 4;
        int blocks = (int)((n4 + 255) / 256);
        cvt_bf16_kernel<<<blocks, 256, 0, stream>>>(agg, aggb, n4);
    }
    // 4) wtb = bf16(W^T)
    {
        dim3 grid(NDIM / 32, NDIM / 32);
        transpose_bf16_kernel<<<grid, 256, 0, stream>>>(W, wtb);
    }
    // 5) ht = relu(aggb @ W)^T      (4096 wave-tiles of 64x64 -> 512 blocks x 8 waves)
    {
        int tiles = (NDIM / 64) * (NDIM / 64);
        gemm1_wmma_kernel<<<tiles / 8, 256, 0, stream>>>(aggb, wtb, ht);
    }
    // 6) out = sigmoid(ht ht^T)     (triangular: 2080 wave-tiles -> 260 blocks x 8 waves)
    {
        int T = NDIM / 64;
        int pairs = T * (T + 1) / 2;               // 2080, divisible by 8
        gemm2_wmma_kernel<<<pairs / 8, 256, 0, stream>>>(ht, (float*)d_out);
    }
}
